// LongformerForBinaryClassification_25752623907087
// MI455X (gfx1250) — matrix operand order repaired and checked by reading, not compiled, not run on hardware
//
#include <hip/hip_runtime.h>

#define S_LEN   2048
#define BATCH   2
#define HIDDIM  768
#define NHEAD   12
#define HD      64
#define NLAYER  4
#define FFDIM   3072
#define WIN     256
#define NCHUNK  (S_LEN / WIN)
#define MROWS   (BATCH * S_LEN)

typedef __bf16 bf16;
typedef __attribute__((ext_vector_type(16))) __bf16 v16bf;
typedef __attribute__((ext_vector_type(8)))  float  v8f;
typedef int v4i __attribute__((__vector_size__(16)));
typedef __attribute__((address_space(1))) v4i gv4i;
typedef __attribute__((address_space(3))) v4i lv4i;

union Frag32B { v16bf v; uint4 q[2]; };
union Frag16B { uint4 q; bf16 e[8]; };

// ---- CDNA5 async global->LDS copy (ASYNCcnt-tracked DMA path) --------------
#if defined(__has_builtin)
#if __has_builtin(__builtin_amdgcn_global_load_async_to_lds_b128) && \
    __has_builtin(__builtin_amdgcn_s_wait_asynccnt)
#define HAVE_ASYNC_LDS 1
#endif
#endif
#ifndef HAVE_ASYNC_LDS
#define HAVE_ASYNC_LDS 0
#endif

__device__ __forceinline__ void lds_cp_b128(const bf16* g, bf16* l) {
#if HAVE_ASYNC_LDS
  __builtin_amdgcn_global_load_async_to_lds_b128(
      (gv4i*)(unsigned long long)g,
      (lv4i*)(unsigned)(unsigned long long)l,
      0, 0);
#else
  *(uint4*)l = *(const uint4*)g;
#endif
}

__device__ __forceinline__ void async_wait_all() {
#if HAVE_ASYNC_LDS
  __builtin_amdgcn_s_wait_asynccnt(0);
#endif
}

__device__ __forceinline__ bf16 f2bf(float f) {
  unsigned u = __builtin_bit_cast(unsigned, f);
  unsigned r = (u + 0x7FFFu + ((u >> 16) & 1u)) >> 16;
  unsigned short h = (unsigned short)r;
  return __builtin_bit_cast(bf16, h);
}

__device__ __forceinline__ v8f wmma_bf16(v16bf a, v16bf b, v8f c) {
  // D = A(16x32 bf16) * B(32x16 bf16) + C(16x16 f32), wave32
  return __builtin_amdgcn_wmma_f32_16x16x32_bf16(false, a, false, b, (short)0, c,
                                                 false, false);
}

__device__ __forceinline__ float gelu_exact(float v) {
  return 0.5f * v * (1.0f + erff(v * 0.70710678118654752f));
}

// ---------------------------------------------------------------------------
// fp32 [K,N] -> bf16 transposed [N,K], 64x64 tile through LDS (coalesced both
// sides). Grid: (N/64, K/64).
// ---------------------------------------------------------------------------
__global__ __launch_bounds__(256) void cvt_t_kernel(const float* __restrict__ in,
                                                    bf16* __restrict__ out,
                                                    int K, int N) {
  __shared__ bf16 t[64][72];
  const int n0 = blockIdx.x * 64;
  const int k0 = blockIdx.y * 64;
  const int tid = threadIdx.x;
  const int rr = tid >> 2;          // 0..63
  const int cc = (tid & 3) * 16;    // 0,16,32,48
  const float* src = in + (size_t)(k0 + rr) * N + n0 + cc;
#pragma unroll
  for (int j = 0; j < 16; j++) t[cc + j][rr] = f2bf(src[j]);
  __syncthreads();
  bf16* dst = out + (size_t)(n0 + rr) * K + k0 + cc;
  *(uint4*)(dst + 0) = *(const uint4*)&t[rr][cc];
  *(uint4*)(dst + 8) = *(const uint4*)&t[rr][cc + 8];
}

// ---------------------------------------------------------------------------
// RoBERTa position ids: pos = cumsum(mask)*mask + 1
// ---------------------------------------------------------------------------
__global__ void posids_kernel(const int* __restrict__ amask, int* __restrict__ pos) {
  int b = blockIdx.x;
  if (threadIdx.x == 0) {
    int run = 0;
    for (int s = 0; s < S_LEN; s++) {
      int m = amask[(size_t)b * S_LEN + s];
      run += m;
      pos[(size_t)b * S_LEN + s] = run * m + 1;
    }
  }
}

// ---------------------------------------------------------------------------
// Embedding gather: word + pos + tokentype -> fp32 [MROWS, HID]
// ---------------------------------------------------------------------------
__global__ __launch_bounds__(256) void embed_kernel(const int* __restrict__ ids,
                                                    const int* __restrict__ pos,
                                                    const float* __restrict__ we,
                                                    const float* __restrict__ pe,
                                                    const float* __restrict__ tt,
                                                    float* __restrict__ out) {
  int row = blockIdx.x;
  int id = ids[row];
  int p = pos[row];
  int tid = threadIdx.x;
#pragma unroll
  for (int i = 0; i < 3; i++) {
    int c = tid + i * 256;
    out[(size_t)row * HIDDIM + c] =
        we[(size_t)id * HIDDIM + c] + pe[(size_t)p * HIDDIM + c] + tt[c];
  }
}

// ---------------------------------------------------------------------------
// LayerNorm over last dim (768). y = LN(a + r); writes fp32 x and bf16 mirror.
// ---------------------------------------------------------------------------
__global__ __launch_bounds__(256) void ln_kernel(const float* __restrict__ a,
                                                 const float* __restrict__ r,
                                                 const float* __restrict__ gamma,
                                                 const float* __restrict__ beta,
                                                 float* __restrict__ xo,
                                                 bf16* __restrict__ xbo) {
  int row = blockIdx.x;
  int tid = threadIdx.x;
  int lane = tid & 31, wid = tid >> 5;
  float vals[3];
  float s = 0.f, s2 = 0.f;
#pragma unroll
  for (int i = 0; i < 3; i++) {
    size_t idx = (size_t)row * HIDDIM + tid + i * 256;
    float v = a[idx] + (r ? r[idx] : 0.0f);
    vals[i] = v;
    s += v;
    s2 += v * v;
  }
#pragma unroll
  for (int off = 16; off > 0; off >>= 1) {
    s += __shfl_down(s, off);
    s2 += __shfl_down(s2, off);
  }
  __shared__ float rs[8], rs2[8];
  if (lane == 0) { rs[wid] = s; rs2[wid] = s2; }
  __syncthreads();
  float ts = 0.f, ts2 = 0.f;
#pragma unroll
  for (int w = 0; w < 8; w++) { ts += rs[w]; ts2 += rs2[w]; }
  float mean = ts * (1.0f / HIDDIM);
  float var = ts2 * (1.0f / HIDDIM) - mean * mean;
  float inv = rsqrtf(var + 1e-5f);
#pragma unroll
  for (int i = 0; i < 3; i++) {
    int c = tid + i * 256;
    size_t idx = (size_t)row * HIDDIM + c;
    float y = (vals[i] - mean) * inv * gamma[c] + beta[c];
    xo[idx] = y;
    xbo[idx] = f2bf(y);
  }
}

// ---------------------------------------------------------------------------
// GEMM: C[M,N] = A[M,K](bf16) * Bt[N,K](bf16, pre-transposed) + bias
// Block tile 128x128, BK=32, 8 waves, each wave 64x32 = 4x2 WMMA tiles.
// Double-buffered LDS; async global->LDS staging overlaps with WMMA.
// act: 0 = none, 1 = exact GELU. Cf (fp32) / Cb (bf16) optional outputs.
// ---------------------------------------------------------------------------
__global__ __launch_bounds__(256) void gemm_bf16_kernel(
    const bf16* __restrict__ A, const bf16* __restrict__ Bt,
    const float* __restrict__ bias, float* __restrict__ Cf,
    bf16* __restrict__ Cb, int M, int N, int K, int act) {
  __shared__ bf16 As[2][128][40];
  __shared__ bf16 Bs[2][128][40];
  const int tid = threadIdx.x;
  const int lane = tid & 31, wid = tid >> 5;
  const int half = lane >> 4, l16 = lane & 15;
  const int bm = blockIdx.y, bn = blockIdx.x;
  const int wave_m = wid >> 2;   // 0..1 -> rows wave_m*64
  const int wave_n = wid & 3;    // 0..3 -> cols wave_n*32

  v8f acc[4][2];
  const v8f vz = {0.f, 0.f, 0.f, 0.f, 0.f, 0.f, 0.f, 0.f};
#pragma unroll
  for (int mt = 0; mt < 4; mt++)
#pragma unroll
    for (int nt = 0; nt < 2; nt++) acc[mt][nt] = vz;

  auto stage = [&](int k0, int buf) {
#pragma unroll
    for (int i = 0; i < 2; i++) {
      int idx = tid + i * 256;   // 0..511 chunks of 16B
      int row = idx >> 2;
      int ch = idx & 3;
      lds_cp_b128(A + (size_t)(bm * 128 + row) * K + k0 + ch * 8,
                  &As[buf][row][ch * 8]);
      lds_cp_b128(Bt + (size_t)(bn * 128 + row) * K + k0 + ch * 8,
                  &Bs[buf][row][ch * 8]);
    }
  };

  const int nk = K >> 5;
  stage(0, 0);
  for (int kt = 0; kt < nk; kt++) {
    const int buf = kt & 1;
    async_wait_all();    // this wave's tile-kt copies have landed in LDS
    __syncthreads();     // all waves landed; prev reads of buf^1 are done
    if (kt + 1 < nk) stage((kt + 1) << 5, buf ^ 1);  // overlaps with WMMAs

    Frag32B af[4], bfr[2];
#pragma unroll
    for (int mt = 0; mt < 4; mt++) {
      int rr = wave_m * 64 + mt * 16 + l16;
      af[mt].q[0] = *(const uint4*)&As[buf][rr][half * 8];
      af[mt].q[1] = *(const uint4*)&As[buf][rr][16 + half * 8];
    }
#pragma unroll
    for (int nt = 0; nt < 2; nt++) {
      int cc = wave_n * 32 + nt * 16 + l16;
      bfr[nt].q[0] = *(const uint4*)&Bs[buf][cc][half * 8];
      bfr[nt].q[1] = *(const uint4*)&Bs[buf][cc][16 + half * 8];
    }
#pragma unroll
    for (int mt = 0; mt < 4; mt++)
#pragma unroll
      for (int nt = 0; nt < 2; nt++)
        acc[mt][nt] = wmma_bf16(af[mt].v, bfr[nt].v, acc[mt][nt]);
  }

#pragma unroll
  for (int mt = 0; mt < 4; mt++)
#pragma unroll
    for (int nt = 0; nt < 2; nt++)
#pragma unroll
      for (int r = 0; r < 8; r++) {
        int gm = bm * 128 + wave_m * 64 + mt * 16 + r + half * 8;
        int gn = bn * 128 + wave_n * 32 + nt * 16 + l16;
        float v = acc[mt][nt][r] + bias[gn];
        if (act == 1) v = gelu_exact(v);
        size_t oidx = (size_t)gm * N + gn;
        if (Cf) Cf[oidx] = v;
        if (Cb) Cb[oidx] = f2bf(v);
      }
}

// ---------------------------------------------------------------------------
// Banded (sliding-window) flash attention.
// One block per (b, head, 256-query chunk). 16 waves, each owns 16 q-rows.
// Key span = [c*W - W, c*W + 2W), iterated in 12 tiles of 64 keys.
// ---------------------------------------------------------------------------
__global__ __launch_bounds__(512) void attn_kernel(const bf16* __restrict__ Q,
                                                   const bf16* __restrict__ Kb,
                                                   const bf16* __restrict__ Vb,
                                                   const int* __restrict__ amask,
                                                   bf16* __restrict__ O) {
  __shared__ bf16 Ks[64][72];        // [key j][d]   (= B-layout for Q*K^T)
  __shared__ bf16 Vs[64][72];        // [d][key j]   (= B-layout for P*V)
  __shared__ bf16 Ps[16][16][72];    // per-wave P exchange [wave][row][key]

  const int bid = blockIdx.x;
  const int c = bid % NCHUNK;
  const int h = (bid / NCHUNK) % NHEAD;
  const int b = bid / (NCHUNK * NHEAD);
  const int tid = threadIdx.x;
  const int lane = tid & 31, wid = tid >> 5;
  const int half = lane >> 4, l16 = lane & 15;

  // Load Q fragments (16 rows x 64 HD = 2 K-steps) straight from global.
  v16bf qf[2];
  {
    int sq = c * WIN + wid * 16 + l16;
    const bf16* qrow = Q + ((size_t)b * S_LEN + sq) * HIDDIM + h * HD;
#pragma unroll
    for (int ks = 0; ks < 2; ks++) {
      Frag32B f;
      f.q[0] = *(const uint4*)(qrow + ks * 32 + half * 8);
      f.q[1] = *(const uint4*)(qrow + ks * 32 + 16 + half * 8);
      qf[ks] = f.v;
    }
  }

  v8f oc[4];
  const v8f vz = {0.f, 0.f, 0.f, 0.f, 0.f, 0.f, 0.f, 0.f};
#pragma unroll
  for (int nt = 0; nt < 4; nt++) oc[nt] = vz;
  float mrun[8], lrun[8];
#pragma unroll
  for (int r = 0; r < 8; r++) { mrun[r] = -1e30f; lrun[r] = 0.f; }

  for (int t = 0; t < 12; t++) {
    int span0 = c * WIN - WIN + t * 64;
    __syncthreads();   // everyone done reading previous K/V tile
    {  // stage K tile (straight copy: K^T columns == K rows) -- async DMA
      int j = tid >> 3, d0 = (tid & 7) * 8;
      int kp = span0 + j;
      if (kp >= 0 && kp < S_LEN) {
        lds_cp_b128(Kb + ((size_t)b * S_LEN + kp) * HIDDIM + h * HD + d0,
                    &Ks[j][d0]);
      } else {
        uint4 z; z.x = z.y = z.z = z.w = 0;
        *(uint4*)&Ks[j][d0] = z;
      }
    }
    {  // stage V tile transposed: Vs[d][j] (scatter, VGPR path)
      int jv = tid & 63, dv0 = (tid >> 6) * 8;
      int kp = span0 + jv;
      Frag16B f;
      f.q.x = f.q.y = f.q.z = f.q.w = 0;
      if (kp >= 0 && kp < S_LEN)
        f.q = *(const uint4*)(Vb + ((size_t)b * S_LEN + kp) * HIDDIM + h * HD + dv0);
#pragma unroll
      for (int i = 0; i < 8; i++) Vs[dv0 + i][jv] = f.e[i];
    }
    async_wait_all();
    __syncthreads();

    // scores S = Q * K^T  (16 x 64)
    v8f sc[4];
#pragma unroll
    for (int nt = 0; nt < 4; nt++) sc[nt] = vz;
#pragma unroll
    for (int ks = 0; ks < 2; ks++) {
#pragma unroll
      for (int nt = 0; nt < 4; nt++) {
        Frag32B bfK;
        bfK.q[0] = *(const uint4*)&Ks[nt * 16 + l16][ks * 32 + half * 8];
        bfK.q[1] = *(const uint4*)&Ks[nt * 16 + l16][ks * 32 + 16 + half * 8];
        sc[nt] = wmma_bf16(qf[ks], bfK.v, sc[nt]);
      }
    }

    // mask + scale + tile row max
    float rowmax[8];
#pragma unroll
    for (int r = 0; r < 8; r++) rowmax[r] = -1e30f;
#pragma unroll
    for (int nt = 0; nt < 4; nt++) {
      int kg = span0 + nt * 16 + l16;
      bool kin = (kg >= 0) && (kg < S_LEN);
      int kidx = kin ? kg : 0;
      bool kvalid = kin && (amask[(size_t)b * S_LEN + kidx] != 0);
#pragma unroll
      for (int r = 0; r < 8; r++) {
        int qpos = c * WIN + wid * 16 + r + half * 8;
        int rel = kg - qpos;
        float sv = (kvalid && rel >= -WIN && rel <= WIN) ? sc[nt][r] * 0.125f
                                                         : -1e9f;
        sc[nt][r] = sv;
        rowmax[r] = fmaxf(rowmax[r], sv);
      }
    }
#pragma unroll
    for (int off = 1; off < 16; off <<= 1)
#pragma unroll
      for (int r = 0; r < 8; r++)
        rowmax[r] = fmaxf(rowmax[r], __shfl_xor(rowmax[r], off));

    float alpha[8], rowsum[8];
#pragma unroll
    for (int r = 0; r < 8; r++) {
      float mn = fmaxf(mrun[r], rowmax[r]);
      alpha[r] = __expf(mrun[r] - mn);
      mrun[r] = mn;
      rowsum[r] = 0.f;
    }
#pragma unroll
    for (int nt = 0; nt < 4; nt++)
#pragma unroll
      for (int r = 0; r < 8; r++) {
        float p = __expf(sc[nt][r] - mrun[r]);
        rowsum[r] += p;
        Ps[wid][r + half * 8][nt * 16 + l16] = f2bf(p);
      }
#pragma unroll
    for (int off = 1; off < 16; off <<= 1)
#pragma unroll
      for (int r = 0; r < 8; r++) rowsum[r] += __shfl_xor(rowsum[r], off);
#pragma unroll
    for (int r = 0; r < 8; r++) lrun[r] = lrun[r] * alpha[r] + rowsum[r];
#pragma unroll
    for (int nt = 0; nt < 4; nt++)
#pragma unroll
      for (int r = 0; r < 8; r++) oc[nt][r] *= alpha[r];

    // O += P * V   (contraction over 64 keys = 2 WMMA K-steps)
#pragma unroll
    for (int ks = 0; ks < 2; ks++) {
      Frag32B pa;
      pa.q[0] = *(const uint4*)&Ps[wid][l16][ks * 32 + half * 8];
      pa.q[1] = *(const uint4*)&Ps[wid][l16][ks * 32 + 16 + half * 8];
#pragma unroll
      for (int nt = 0; nt < 4; nt++) {
        Frag32B vb2;
        vb2.q[0] = *(const uint4*)&Vs[nt * 16 + l16][ks * 32 + half * 8];
        vb2.q[1] = *(const uint4*)&Vs[nt * 16 + l16][ks * 32 + 16 + half * 8];
        oc[nt] = wmma_bf16(pa.v, vb2.v, oc[nt]);
      }
    }
  }

  // normalize and store (bf16, [B,S,HID] with heads concatenated)
#pragma unroll
  for (int nt = 0; nt < 4; nt++)
#pragma unroll
    for (int r = 0; r < 8; r++) {
      float val = oc[nt][r] / fmaxf(lrun[r], 1e-20f);
      int s = c * WIN + wid * 16 + r + half * 8;
      int col = h * HD + nt * 16 + l16;
      O[((size_t)b * S_LEN + s) * HIDDIM + col] = f2bf(val);
    }
}

// ---------------------------------------------------------------------------
// Pooler + classifier: logits[b] = tanh(x[b,0,:] @ pool_w + pool_b) @ cls_w + cls_b
// ---------------------------------------------------------------------------
__global__ __launch_bounds__(256) void pool_kernel(const float* __restrict__ x,
                                                   const float* __restrict__ pw,
                                                   const float* __restrict__ pb,
                                                   const float* __restrict__ cw,
                                                   const float* __restrict__ cb,
                                                   float* __restrict__ out) {
  int b = blockIdx.x;
  int tid = threadIdx.x;
  __shared__ float pooled[HIDDIM];
  const float* xr = x + (size_t)b * S_LEN * HIDDIM;  // row s=0
#pragma unroll
  for (int i = 0; i < 3; i++) {
    int j = tid + i * 256;
    float acc = pb[j];
    for (int k = 0; k < HIDDIM; k++) acc += xr[k] * pw[(size_t)k * HIDDIM + j];
    pooled[j] = tanhf(acc);
  }
  __syncthreads();
  float part = 0.f;
#pragma unroll
  for (int i = 0; i < 3; i++) {
    int j = tid + i * 256;
    part += pooled[j] * cw[j];
  }
#pragma unroll
  for (int off = 16; off > 0; off >>= 1) part += __shfl_down(part, off);
  __shared__ float wsum[8];
  int lane = tid & 31, wid = tid >> 5;
  if (lane == 0) wsum[wid] = part;
  __syncthreads();
  if (tid == 0) {
    float t = cb[0];
#pragma unroll
    for (int w = 0; w < 8; w++) t += wsum[w];
    out[b] = t;
  }
}

// ---------------------------------------------------------------------------
extern "C" void kernel_launch(void* const* d_in, const int* in_sizes, int n_in,
                              void* d_out, int out_size, void* d_ws, size_t ws_size,
                              hipStream_t stream) {
  const int*   ids    = (const int*)d_in[0];
  const int*   amask  = (const int*)d_in[1];
  const float* wemb   = (const float*)d_in[2];
  const float* pemb   = (const float*)d_in[3];
  const float* ttemb  = (const float*)d_in[4];
  const float* elns   = (const float*)d_in[5];
  const float* elnb   = (const float*)d_in[6];
  const float* Wq     = (const float*)d_in[7];
  const float* bq     = (const float*)d_in[8];
  const float* Wk     = (const float*)d_in[9];
  const float* bk     = (const float*)d_in[10];
  const float* Wv     = (const float*)d_in[11];
  const float* bv     = (const float*)d_in[12];
  const float* Wo     = (const float*)d_in[13];
  const float* bo     = (const float*)d_in[14];
  const float* alns   = (const float*)d_in[15];
  const float* alnb   = (const float*)d_in[16];
  const float* Wi     = (const float*)d_in[17];
  const float* bi     = (const float*)d_in[18];
  const float* Wf     = (const float*)d_in[19];
  const float* bfw    = (const float*)d_in[20];
  const float* flns   = (const float*)d_in[21];
  const float* flnb   = (const float*)d_in[22];
  const float* poolw  = (const float*)d_in[23];
  const float* poolb  = (const float*)d_in[24];
  const float* clsw   = (const float*)d_in[25];
  const float* clsb   = (const float*)d_in[26];

  char* ws = (char*)d_ws;
  size_t off = 0;
  auto alloc = [&](size_t bytes) -> char* {
    char* p = ws + off;
    off += (bytes + 255) & ~(size_t)255;
    return p;
  };
  bf16* wqT = (bf16*)alloc((size_t)NLAYER * HIDDIM * HIDDIM * 2);
  bf16* wkT = (bf16*)alloc((size_t)NLAYER * HIDDIM * HIDDIM * 2);
  bf16* wvT = (bf16*)alloc((size_t)NLAYER * HIDDIM * HIDDIM * 2);
  bf16* woT = (bf16*)alloc((size_t)NLAYER * HIDDIM * HIDDIM * 2);
  bf16* wiT = (bf16*)alloc((size_t)NLAYER * HIDDIM * FFDIM * 2);
  bf16* wfT = (bf16*)alloc((size_t)NLAYER * FFDIM * HIDDIM * 2);
  float* x   = (float*)alloc((size_t)MROWS * HIDDIM * 4);
  bf16*  xb  = (bf16*)alloc((size_t)MROWS * HIDDIM * 2);
  bf16*  qb  = (bf16*)alloc((size_t)MROWS * HIDDIM * 2);
  bf16*  kb2 = (bf16*)alloc((size_t)MROWS * HIDDIM * 2);
  bf16*  vb2 = (bf16*)alloc((size_t)MROWS * HIDDIM * 2);
  bf16*  ab  = (bf16*)alloc((size_t)MROWS * HIDDIM * 2);
  bf16*  hb  = (bf16*)alloc((size_t)MROWS * FFDIM * 2);
  float* tmpf = (float*)alloc((size_t)MROWS * HIDDIM * 4);
  int*   posids = (int*)alloc((size_t)MROWS * 4);

  auto cvt = [&](const float* in, bf16* out, int K, int N) {
    dim3 g(N / 64, K / 64);
    cvt_t_kernel<<<g, 256, 0, stream>>>(in, out, K, N);
  };
  for (int l = 0; l < NLAYER; l++) {
    cvt(Wq + (size_t)l * HIDDIM * HIDDIM, wqT + (size_t)l * HIDDIM * HIDDIM, HIDDIM, HIDDIM);
    cvt(Wk + (size_t)l * HIDDIM * HIDDIM, wkT + (size_t)l * HIDDIM * HIDDIM, HIDDIM, HIDDIM);
    cvt(Wv + (size_t)l * HIDDIM * HIDDIM, wvT + (size_t)l * HIDDIM * HIDDIM, HIDDIM, HIDDIM);
    cvt(Wo + (size_t)l * HIDDIM * HIDDIM, woT + (size_t)l * HIDDIM * HIDDIM, HIDDIM, HIDDIM);
    cvt(Wi + (size_t)l * HIDDIM * FFDIM,  wiT + (size_t)l * HIDDIM * FFDIM,  HIDDIM, FFDIM);
    cvt(Wf + (size_t)l * FFDIM * HIDDIM,  wfT + (size_t)l * FFDIM * HIDDIM,  FFDIM, HIDDIM);
  }

  posids_kernel<<<BATCH, 32, 0, stream>>>(amask, posids);
  embed_kernel<<<MROWS, 256, 0, stream>>>(ids, posids, wemb, pemb, ttemb, tmpf);
  ln_kernel<<<MROWS, 256, 0, stream>>>(tmpf, nullptr, elns, elnb, x, xb);

  auto gemm = [&](const bf16* A, const bf16* Bt, const float* bias, float* Cf,
                  bf16* Cb, int M, int N, int K, int act) {
    dim3 g(N / 128, M / 128);
    gemm_bf16_kernel<<<g, 256, 0, stream>>>(A, Bt, bias, Cf, Cb, M, N, K, act);
  };

  for (int l = 0; l < NLAYER; l++) {
    const size_t wofs = (size_t)l * HIDDIM * HIDDIM;
    gemm(xb, wqT + wofs, bq + (size_t)l * HIDDIM, nullptr, qb, MROWS, HIDDIM, HIDDIM, 0);
    gemm(xb, wkT + wofs, bk + (size_t)l * HIDDIM, nullptr, kb2, MROWS, HIDDIM, HIDDIM, 0);
    gemm(xb, wvT + wofs, bv + (size_t)l * HIDDIM, nullptr, vb2, MROWS, HIDDIM, HIDDIM, 0);
    attn_kernel<<<BATCH * NHEAD * NCHUNK, 512, 0, stream>>>(qb, kb2, vb2, amask, ab);
    gemm(ab, woT + wofs, bo + (size_t)l * HIDDIM, tmpf, nullptr, MROWS, HIDDIM, HIDDIM, 0);
    ln_kernel<<<MROWS, 256, 0, stream>>>(x, tmpf, alns + (size_t)l * HIDDIM,
                                         alnb + (size_t)l * HIDDIM, x, xb);
    gemm(xb, wiT + (size_t)l * HIDDIM * FFDIM, bi + (size_t)l * FFDIM, nullptr, hb,
         MROWS, FFDIM, HIDDIM, 1);
    gemm(hb, wfT + (size_t)l * FFDIM * HIDDIM, bfw + (size_t)l * HIDDIM, tmpf, nullptr,
         MROWS, HIDDIM, FFDIM, 0);
    ln_kernel<<<MROWS, 256, 0, stream>>>(x, tmpf, flns + (size_t)l * HIDDIM,
                                         flnb + (size_t)l * HIDDIM, x, xb);
  }

  pool_kernel<<<BATCH, 256, 0, stream>>>(x, poolw, poolb, clsw, clsb, (float*)d_out);
}
